// MultiHeadSelfAttention_41979010351300
// MI455X (gfx1250) — matrix operand order repaired and checked by reading, not compiled, not run on hardware
//
#include <hip/hip_runtime.h>
#include <hip/hip_bf16.h>
#include <math.h>

// ---------- types ----------
typedef __bf16 bf16_t;
typedef __attribute__((ext_vector_type(8)))  __bf16 v8bf;
typedef __attribute__((ext_vector_type(16))) __bf16 v16bf;
typedef __attribute__((ext_vector_type(8)))  float  v8f;

union ABFrag { v16bf v; v8bf h[2]; };

#define N_EMBD 1024
#define N_HEAD 16
#define HEAD_DIM 64
#define SEQ_T 2048
#define BATCH 4
#define ROWS (BATCH * SEQ_T)   // 8192

// ---------- WMMA wrapper ----------
__device__ inline v8f wmma_bf16(v16bf a, v16bf b, v8f c) {
  // D = A(16x32 bf16) * B(32x16 bf16) + C(16x16 f32)
  return __builtin_amdgcn_wmma_f32_16x16x32_bf16(
      /*neg_a=*/false, a, /*neg_b=*/false, b,
      /*c_mod=*/(short)0, c, /*reuse_a=*/false, /*reuse_b=*/false);
}

// WGP-scope prefetch (SCOPE=0): fills WGP$ + GL2 on miss. The __builtin_prefetch
// locality knob only reaches SE/SYS scope, which skips the WGP cache entirely.
__device__ inline void prefetch_wgp(const void* p) {
  asm volatile("global_prefetch_b8 %0, off" :: "v"(p) : "memory");
}

// A fragment (16x32, row-major source, rows = M, contiguous K).
// ISA layout: lanes 0-15 = M, VGPR0-3: K = half*8..+7, VGPR4-7: K = 16+half*8..+7
__device__ inline v16bf load_frag_a(const bf16_t* __restrict__ base, int ld, int lane) {
  const int m = lane & 15, half = lane >> 4;
  const bf16_t* p = base + (size_t)m * ld + half * 8;
  ABFrag f;
  f.h[0] = *(const v8bf*)(p);        // K = half*8 .. half*8+7
  f.h[1] = *(const v8bf*)(p + 16);   // K = 16+half*8 .. +7
  return f.v;
}

// B fragment (32x16): B[k][n]; source is "transposed" row-major [N][K]
// ISA layout: lane n holds column n, VGPR0-3: K = half*8..+7, VGPR4-7: K = 16+half*8..+7
__device__ inline v16bf load_frag_b(const bf16_t* __restrict__ baseT, int ld, int lane) {
  const int n = lane & 15, half = lane >> 4;
  const bf16_t* p = baseT + (size_t)n * ld + half * 8;
  ABFrag f;
  f.h[0] = *(const v8bf*)(p);
  f.h[1] = *(const v8bf*)(p + 16);
  return f.v;
}

// ds_swizzle xor-broadcast within 16-lane half (masks <= 8)
template <int MASK>
__device__ inline float swz_xor(float v) {
  int i = __builtin_amdgcn_ds_swizzle(__float_as_int(v), (MASK << 10) | 0x1f);
  return __int_as_float(i);
}

// ---------- fp32 -> bf16 convert (8 elems / thread) ----------
__global__ void __launch_bounds__(256)
cvt_f32_bf16(const float* __restrict__ src, bf16_t* __restrict__ dst, int n8) {
  int i = blockIdx.x * blockDim.x + threadIdx.x;
  if (i >= n8) return;
  const float4* s = (const float4*)src;
  float4 a = s[2 * i], b = s[2 * i + 1];
  v8bf o;
  o[0] = (bf16_t)a.x; o[1] = (bf16_t)a.y; o[2] = (bf16_t)a.z; o[3] = (bf16_t)a.w;
  o[4] = (bf16_t)b.x; o[5] = (bf16_t)b.y; o[6] = (bf16_t)b.z; o[7] = (bf16_t)b.w;
  ((v8bf*)dst)[i] = o;
}

// ---------- GEMM: Y = A(Mx1024) * Wt^T, Wt stored row-major [1024][1024] ----------
// mode 0: bf16 out (B,H,T,D)      (Q,K)
// mode 1: bf16 out (B,H,D,T)      (V transposed)
// mode 2: f32  out row-major Mx1024 (final projection)
__global__ void __launch_bounds__(256)
gemm_bf16_k(const bf16_t* __restrict__ A, const bf16_t* __restrict__ Wt,
            void* __restrict__ out, int mode) {
  const int lane = threadIdx.x & 31;
  const int wave = threadIdx.x >> 5;
  const int wm = wave & 1, wn = wave >> 1;
  const int mBase = blockIdx.x * 64 + wm * 32;
  const int nBase = blockIdx.y * 256 + wn * 64;

  v8f acc[2][4];
#pragma unroll
  for (int mi = 0; mi < 2; ++mi)
#pragma unroll
    for (int nf = 0; nf < 4; ++nf) acc[mi][nf] = {};

#pragma unroll 4
  for (int k = 0; k < N_EMBD; k += 32) {
    // prefetch the streaming A operand one k-tile ahead into the WGP cache:
    // 32 lanes x one byte-probe per row covers the whole 32x32 next tile
    if (k + 32 < N_EMBD)
      prefetch_wgp(A + (size_t)(mBase + lane) * N_EMBD + k + 32);

    v16bf af[2];
    af[0] = load_frag_a(A + (size_t)mBase * N_EMBD + k, N_EMBD, lane);
    af[1] = load_frag_a(A + (size_t)(mBase + 16) * N_EMBD + k, N_EMBD, lane);
    v16bf bfr[4];
#pragma unroll
    for (int nf = 0; nf < 4; ++nf)
      bfr[nf] = load_frag_b(Wt + (size_t)(nBase + nf * 16) * N_EMBD + k, N_EMBD, lane);
#pragma unroll
    for (int mi = 0; mi < 2; ++mi)
#pragma unroll
      for (int nf = 0; nf < 4; ++nf)
        acc[mi][nf] = wmma_bf16(af[mi], bfr[nf], acc[mi][nf]);
  }

  const int half = lane >> 4, nl = lane & 15;
#pragma unroll
  for (int mi = 0; mi < 2; ++mi)
#pragma unroll
    for (int nf = 0; nf < 4; ++nf)
#pragma unroll
      for (int r = 0; r < 8; ++r) {
        int row = mBase + mi * 16 + r + half * 8;    // token index
        int col = nBase + nf * 16 + nl;              // embed index
        float v = acc[mi][nf][r];
        if (mode == 2) {
          ((float*)out)[(size_t)row * N_EMBD + col] = v;
        } else {
          int b = row >> 11, t = row & (SEQ_T - 1);
          int h = col >> 6, d = col & (HEAD_DIM - 1);
          size_t idx = (mode == 0)
              ? ((size_t)((b * N_HEAD + h) * SEQ_T + t)) * HEAD_DIM + d
              : ((size_t)((b * N_HEAD + h) * HEAD_DIM + d)) * SEQ_T + t;
          ((bf16_t*)out)[idx] = (bf16_t)v;
        }
      }
}

// ---------- Flash attention: one wave per (b, h, 32-query tile) ----------
__global__ void __launch_bounds__(32)
attn_k(const bf16_t* __restrict__ Q, const bf16_t* __restrict__ K,
       const bf16_t* __restrict__ Vt, bf16_t* __restrict__ Aout) {
  const int lane = threadIdx.x;
  const int half = lane >> 4, nl = lane & 15;
  const int qBase = blockIdx.x * 32;
  const int h = blockIdx.y, b = blockIdx.z;

  const size_t headTD = (size_t)(b * N_HEAD + h) * SEQ_T * HEAD_DIM;
  const bf16_t* Qh = Q + headTD;
  const bf16_t* Kh = K + headTD;
  const bf16_t* Vh = Vt + headTD;   // (D, T) layout within head

  __shared__ alignas(64) bf16_t Pl[32 * 32];

  // Q fragments: 2 M-tiles x 2 k-steps (D = 64)
  v16bf qa[2][2];
#pragma unroll
  for (int mb = 0; mb < 2; ++mb)
#pragma unroll
    for (int kk = 0; kk < 2; ++kk)
      qa[mb][kk] = load_frag_a(Qh + (size_t)(qBase + mb * 16) * HEAD_DIM + kk * 32,
                               HEAD_DIM, lane);

  v8f o[2][4];
  float rmax[2][8], rsum[2][8];
#pragma unroll
  for (int mb = 0; mb < 2; ++mb) {
#pragma unroll
    for (int nf = 0; nf < 4; ++nf) o[mb][nf] = {};
#pragma unroll
    for (int r = 0; r < 8; ++r) { rmax[mb][r] = -3.0e38f; rsum[mb][r] = 0.0f; }
  }

  const float SCALE = 0.125f;   // 1/sqrt(64)

  for (int kb = 0; kb <= qBase; kb += 32) {
    // prefetch next key/value tiles (WGP scope) while this tile's softmax runs
    if (kb + 32 <= qBase) {
      prefetch_wgp(Kh + (size_t)(kb + 32 + lane) * HEAD_DIM);
      prefetch_wgp(Vh + (size_t)(lane * 2) * SEQ_T + kb + 32);
    }

    // ---- S = Q * K^T over this 32-key tile ----
    v16bf kf[2][2];
#pragma unroll
    for (int nb = 0; nb < 2; ++nb)
#pragma unroll
      for (int kk = 0; kk < 2; ++kk)
        kf[nb][kk] = load_frag_b(Kh + (size_t)(kb + nb * 16) * HEAD_DIM + kk * 32,
                                 HEAD_DIM, lane);
    v8f s[2][2];
#pragma unroll
    for (int mb = 0; mb < 2; ++mb)
#pragma unroll
      for (int nb = 0; nb < 2; ++nb) {
        v8f c = {};
        c = wmma_bf16(qa[mb][0], kf[nb][0], c);
        c = wmma_bf16(qa[mb][1], kf[nb][1], c);
        s[mb][nb] = c;
      }

    const bool diag = (kb == qBase);

    // ---- online softmax per row ----
#pragma unroll
    for (int mb = 0; mb < 2; ++mb) {
      float corr[8];
#pragma unroll
      for (int r = 0; r < 8; ++r) {
        const int row = qBase + mb * 16 + r + half * 8;
        float s0 = s[mb][0][r] * SCALE;
        float s1 = s[mb][1][r] * SCALE;
        if (diag) {
          if (kb + nl > row)      s0 = -3.0e38f;
          if (kb + 16 + nl > row) s1 = -3.0e38f;
        }
        float v = fmaxf(s0, s1);
        v = fmaxf(v, swz_xor<1>(v));
        v = fmaxf(v, swz_xor<2>(v));
        v = fmaxf(v, swz_xor<4>(v));
        v = fmaxf(v, swz_xor<8>(v));
        float nm = fmaxf(rmax[mb][r], v);
        float c0 = __expf(rmax[mb][r] - nm);
        rmax[mb][r] = nm;
        corr[r] = c0;
        float p0 = __expf(s0 - nm);
        float p1 = __expf(s1 - nm);
        float ts = p0 + p1;
        ts += swz_xor<1>(ts);
        ts += swz_xor<2>(ts);
        ts += swz_xor<4>(ts);
        ts += swz_xor<8>(ts);
        rsum[mb][r] = rsum[mb][r] * c0 + ts;
        // stage P (C layout -> LDS, row-major 32x32)
        const int lrow = mb * 16 + r + half * 8;
        Pl[lrow * 32 + nl]      = (bf16_t)p0;
        Pl[lrow * 32 + 16 + nl] = (bf16_t)p1;
      }
#pragma unroll
      for (int nf = 0; nf < 4; ++nf)
#pragma unroll
        for (int r = 0; r < 8; ++r) o[mb][nf][r] *= corr[r];
    }

    asm volatile("s_wait_dscnt 0" ::: "memory");

    // ---- reload P as A fragments (K-dim = 32 keys, one WMMA step) ----
    v16bf pa[2];
#pragma unroll
    for (int mb = 0; mb < 2; ++mb) {
      const bf16_t* p = &Pl[(mb * 16 + nl) * 32 + half * 8];
      ABFrag f;
      f.h[0] = *(const v8bf*)(p);
      f.h[1] = *(const v8bf*)(p + 16);
      pa[mb] = f.v;
    }

    // ---- O += P * V (V transposed: contiguous key dim per d-column) ----
#pragma unroll
    for (int nf = 0; nf < 4; ++nf) {
      v16bf vf = load_frag_b(Vh + (size_t)(nf * 16) * SEQ_T + kb, SEQ_T, lane);
      o[0][nf] = wmma_bf16(pa[0], vf, o[0][nf]);
      o[1][nf] = wmma_bf16(pa[1], vf, o[1][nf]);
    }
  }

  // ---- normalize and store attn output as bf16 (B, T, E) ----
#pragma unroll
  for (int mb = 0; mb < 2; ++mb) {
    float inv[8];
#pragma unroll
    for (int r = 0; r < 8; ++r) inv[r] = 1.0f / rsum[mb][r];
#pragma unroll
    for (int nf = 0; nf < 4; ++nf)
#pragma unroll
      for (int r = 0; r < 8; ++r) {
        int row = qBase + mb * 16 + r + half * 8;
        int col = h * HEAD_DIM + nf * 16 + nl;
        Aout[((size_t)b * SEQ_T + row) * N_EMBD + col] = (bf16_t)(o[mb][nf][r] * inv[r]);
      }
  }
}

// ---------- host launch ----------
extern "C" void kernel_launch(void* const* d_in, const int* in_sizes, int n_in,
                              void* d_out, int out_size, void* d_ws, size_t ws_size,
                              hipStream_t stream) {
  const float* x  = (const float*)d_in[0];
  const float* Wq = (const float*)d_in[1];
  const float* Wk = (const float*)d_in[2];
  const float* Wv = (const float*)d_in[3];
  const float* Wo = (const float*)d_in[4];
  float* out = (float*)d_out;

  const size_t XN = (size_t)ROWS * N_EMBD;      // 8388608
  const size_t WN = (size_t)N_EMBD * N_EMBD;    // 1048576

  bf16_t* xb  = (bf16_t*)d_ws;
  bf16_t* wqb = xb + XN;
  bf16_t* wkb = wqb + WN;
  bf16_t* wvb = wkb + WN;
  bf16_t* wob = wvb + WN;
  bf16_t* Qb  = wob + WN;       // (B,H,T,D)
  bf16_t* Kb  = Qb + XN;        // (B,H,T,D)
  bf16_t* Vtb = Kb + XN;        // (B,H,D,T)
  bf16_t* Ab  = Vtb + XN;       // (B,T,E)

  // fp32 -> bf16
  cvt_f32_bf16<<<(int)(XN / 8 + 255) / 256, 256, 0, stream>>>(x, xb, (int)(XN / 8));
  cvt_f32_bf16<<<(int)(WN / 8 + 255) / 256, 256, 0, stream>>>(Wq, wqb, (int)(WN / 8));
  cvt_f32_bf16<<<(int)(WN / 8 + 255) / 256, 256, 0, stream>>>(Wk, wkb, (int)(WN / 8));
  cvt_f32_bf16<<<(int)(WN / 8 + 255) / 256, 256, 0, stream>>>(Wv, wvb, (int)(WN / 8));
  cvt_f32_bf16<<<(int)(WN / 8 + 255) / 256, 256, 0, stream>>>(Wo, wob, (int)(WN / 8));

  // QKV projections (WMMA GEMMs)
  dim3 ggrid(ROWS / 64, N_EMBD / 256);
  gemm_bf16_k<<<ggrid, 256, 0, stream>>>(xb, wqb, Qb, 0);
  gemm_bf16_k<<<ggrid, 256, 0, stream>>>(xb, wkb, Kb, 0);
  gemm_bf16_k<<<ggrid, 256, 0, stream>>>(xb, wvb, Vtb, 1);

  // causal flash attention
  dim3 agrid(SEQ_T / 32, N_HEAD, BATCH);
  attn_k<<<agrid, 32, 0, stream>>>(Qb, Kb, Vtb, Ab);

  // output projection -> f32
  gemm_bf16_k<<<ggrid, 256, 0, stream>>>(Ab, wob, (void*)out, 2);
}
